// GraphAttentionLayer_38809324486681
// MI455X (gfx1250) — matrix operand order repaired and checked by reading, not compiled, not run on hardware
//
#include <hip/hip_runtime.h>

#define N 8192
#define IN_F 256
#define OUT_F 64
#define ALPHA 0.2f
#define LOG2E 1.4426950408889634f
// NEG_INF * log2(e): huge-negative sentinel in the exp2 domain (finite, like ref)
#define NEGL2 -1.2984225e16f

typedef __attribute__((ext_vector_type(16))) _Float16 v16h;
typedef __attribute__((ext_vector_type(2)))  __fp16   v2hf;   // cvt_pkrtz result type
typedef __attribute__((ext_vector_type(8)))  float    v8f;
typedef __attribute__((ext_vector_type(4)))  float    v4f;
typedef __attribute__((ext_vector_type(4)))  int      v4i;

union AFrag { v16h v; v2hf h2[8]; };

// ---------------------------------------------------------------------------
// Kernel 1: Wh = h @ W (f32), emit WhT (f16, [64][8192]) and attention logits
// src/dst PRE-SCALED by log2(e) so kernel 2 runs softmax in the exp2 domain.
// ---------------------------------------------------------------------------
__global__ __launch_bounds__(64) void gat_wh_kernel(
    const float* __restrict__ h, const float* __restrict__ W,
    const float* __restrict__ a, float* __restrict__ src,
    float* __restrict__ dst, _Float16* __restrict__ WhT)
{
    __shared__ float hrow[IN_F];
    __shared__ float whrow[OUT_F];
    const int row = blockIdx.x;
    const int tid = threadIdx.x;

    for (int k = tid; k < IN_F; k += 64) hrow[k] = h[(size_t)row * IN_F + k];
    __syncthreads();

    float acc = 0.0f;
    #pragma clang loop unroll_count(8)
    for (int k = 0; k < IN_F; ++k) acc += hrow[k] * W[k * OUT_F + tid];

    whrow[tid] = acc;
    WhT[(size_t)tid * N + row] = (_Float16)acc;   // f16 transpose for WMMA B
    __syncthreads();

    if (tid < 2) {
        const float* av = a + tid * OUT_F;
        float s = 0.0f;
        #pragma clang loop unroll(full)
        for (int f = 0; f < OUT_F; ++f) s += whrow[f] * av[f];
        s *= LOG2E;                               // exp2-domain pre-scale
        if (tid == 0) src[row] = s; else dst[row] = s;
    }
}

// ---------------------------------------------------------------------------
// Kernel 2: fused masked-softmax attention, flash-style online softmax in the
// exp2 domain.  One wave32 owns a 16-row tile; K=64 columns per iteration ->
// 8x v_wmma_f32_16x16x32_f16.  Accumulator rescale is branch-skipped unless
// the running row-max actually changed (rare: ~log(#chunks) times).
// ---------------------------------------------------------------------------
__global__ __launch_bounds__(64) void gat_attn_kernel(
    const int* __restrict__ adj, const float* __restrict__ src,
    const float* __restrict__ dst, const _Float16* __restrict__ WhT,
    float* __restrict__ out)
{
    const int lane  = threadIdx.x & 31;
    const int wid   = threadIdx.x >> 5;
    const int tile  = blockIdx.x * 2 + wid;   // 0..511
    const int i0    = tile * 16;
    const int row   = lane & 15;              // A-layout row owned by this lane
    const int half  = lane >> 4;              // 0: K 0-15, 1: K 16-31

    const float srcv = src[i0 + row];
    const int* adjrow = adj + (size_t)(i0 + row) * N;
    const _Float16* bb = WhT + (size_t)row * N + half * 16;

    v8f acc0 = {}, acc1 = {}, acc2 = {}, acc3 = {};
    float m = NEGL2;
    float l = 0.0f;

    // compute 16 masked exp2-domain logits for a 32-col sub-chunk at jb
    auto load_e = [&](int jb, float* e) {
        const int c0 = jb + half * 16;
        const int c1 = c0 + 8;
        v4i a0 = __builtin_nontemporal_load((const v4i*)(adjrow + c0));
        v4i a1 = __builtin_nontemporal_load((const v4i*)(adjrow + c0 + 4));
        v4i a2 = __builtin_nontemporal_load((const v4i*)(adjrow + c1));
        v4i a3 = __builtin_nontemporal_load((const v4i*)(adjrow + c1 + 4));
        v4f d0 = *(const v4f*)(dst + c0);
        v4f d1 = *(const v4f*)(dst + c0 + 4);
        v4f d2 = *(const v4f*)(dst + c1);
        v4f d3 = *(const v4f*)(dst + c1 + 4);
        #pragma clang loop unroll(full)
        for (int i = 0; i < 4; ++i) {
            float x0 = srcv + d0[i]; x0 = fmaxf(x0, ALPHA * x0);  // leaky, a<1
            float x1 = srcv + d1[i]; x1 = fmaxf(x1, ALPHA * x1);
            float x2 = srcv + d2[i]; x2 = fmaxf(x2, ALPHA * x2);
            float x3 = srcv + d3[i]; x3 = fmaxf(x3, ALPHA * x3);
            e[i]      = (a0[i] > 0) ? x0 : NEGL2;
            e[i + 4]  = (a1[i] > 0) ? x1 : NEGL2;
            e[i + 8]  = (a2[i] > 0) ? x2 : NEGL2;
            e[i + 12] = (a3[i] > 0) ? x3 : NEGL2;
        }
    };

    for (int j = 0; j < N; j += 64) {
        float e0[16], e1[16];
        load_e(j, e0);
        load_e(j + 32, e1);

        // row max across 64 cols (row split across lane pair L, L+16)
        float lmax = e0[0];
        #pragma clang loop unroll(full)
        for (int i = 1; i < 16; ++i) lmax = fmaxf(lmax, e0[i]);
        #pragma clang loop unroll(full)
        for (int i = 0; i < 16; ++i) lmax = fmaxf(lmax, e1[i]);
        lmax = fmaxf(lmax, __shfl_xor(lmax, 16, 32));
        const float newm = fmaxf(m, lmax);

        // rescale only when some row's max actually increased (rare)
        if (__any(newm > m)) {
            const float scale = __builtin_amdgcn_exp2f(m - newm);
            l *= scale;
            float sc[8];
            #pragma clang loop unroll(full)
            for (int r = 0; r < 8; ++r) sc[r] = __shfl(scale, half * 8 + r, 32);
            #pragma clang loop unroll(full)
            for (int r = 0; r < 8; ++r) {
                acc0[r] *= sc[r]; acc1[r] *= sc[r];
                acc2[r] *= sc[r]; acc3[r] *= sc[r];
            }
        }
        m = newm;

        // probabilities + row sum
        float p0[16], p1[16];
        float s0 = 0.0f, s1 = 0.0f;
        #pragma clang loop unroll(full)
        for (int i = 0; i < 16; ++i) {
            p0[i] = __builtin_amdgcn_exp2f(e0[i] - newm); s0 += p0[i];
            p1[i] = __builtin_amdgcn_exp2f(e1[i] - newm); s1 += p1[i];
        }
        float lsum = s0 + s1;
        lsum += __shfl_xor(lsum, 16, 32);
        l += lsum;

        // pack A fragments with v_cvt_pk_rtz_f16_f32
        AFrag A0, A1;
        #pragma clang loop unroll(full)
        for (int g = 0; g < 8; ++g) {
            A0.h2[g] = __builtin_amdgcn_cvt_pkrtz(p0[2 * g], p0[2 * g + 1]);
            A1.h2[g] = __builtin_amdgcn_cvt_pkrtz(p1[2 * g], p1[2 * g + 1]);
        }

        // B fragments: col = row + 16*t, K = j(+32) + 16*half + [0..15]
        const _Float16* b0 = bb + j;
        v16h B00 = *(const v16h*)(b0);
        v16h B01 = *(const v16h*)(b0 + 16 * N);
        v16h B02 = *(const v16h*)(b0 + 32 * N);
        v16h B03 = *(const v16h*)(b0 + 48 * N);
        acc0 = __builtin_amdgcn_wmma_f32_16x16x32_f16(false, A0.v, false, B00, (short)0, acc0, false, false);
        acc1 = __builtin_amdgcn_wmma_f32_16x16x32_f16(false, A0.v, false, B01, (short)0, acc1, false, false);
        acc2 = __builtin_amdgcn_wmma_f32_16x16x32_f16(false, A0.v, false, B02, (short)0, acc2, false, false);
        acc3 = __builtin_amdgcn_wmma_f32_16x16x32_f16(false, A0.v, false, B03, (short)0, acc3, false, false);

        const _Float16* b1 = b0 + 32;
        v16h B10 = *(const v16h*)(b1);
        v16h B11 = *(const v16h*)(b1 + 16 * N);
        v16h B12 = *(const v16h*)(b1 + 32 * N);
        v16h B13 = *(const v16h*)(b1 + 48 * N);
        acc0 = __builtin_amdgcn_wmma_f32_16x16x32_f16(false, A1.v, false, B10, (short)0, acc0, false, false);
        acc1 = __builtin_amdgcn_wmma_f32_16x16x32_f16(false, A1.v, false, B11, (short)0, acc1, false, false);
        acc2 = __builtin_amdgcn_wmma_f32_16x16x32_f16(false, A1.v, false, B12, (short)0, acc2, false, false);
        acc3 = __builtin_amdgcn_wmma_f32_16x16x32_f16(false, A1.v, false, B13, (short)0, acc3, false, false);
    }

    // normalize by row sums and store (C layout: vgpr r -> M = r + 8*half)
    const float linv = __builtin_amdgcn_rcpf(l);
    float li[8];
    #pragma clang loop unroll(full)
    for (int r = 0; r < 8; ++r) li[r] = __shfl(linv, half * 8 + r, 32);

    #pragma clang loop unroll(full)
    for (int r = 0; r < 8; ++r) {
        const int orow = i0 + half * 8 + r;
        float* po = out + (size_t)orow * OUT_F + row;
        po[0]  = acc0[r] * li[r];
        po[16] = acc1[r] * li[r];
        po[32] = acc2[r] * li[r];
        po[48] = acc3[r] * li[r];
    }
}

// ---------------------------------------------------------------------------
extern "C" void kernel_launch(void* const* d_in, const int* in_sizes, int n_in,
                              void* d_out, int out_size, void* d_ws, size_t ws_size,
                              hipStream_t stream) {
    (void)in_sizes; (void)n_in; (void)out_size; (void)ws_size;
    const float* h   = (const float*)d_in[0];
    const float* W   = (const float*)d_in[1];
    const float* a   = (const float*)d_in[2];
    const int*   adj = (const int*)d_in[3];
    float* out = (float*)d_out;

    char* ws = (char*)d_ws;
    float*     src = (float*)(ws);                    // 32 KB
    float*     dst = (float*)(ws + 32 * 1024);        // 32 KB
    _Float16*  WhT = (_Float16*)(ws + 64 * 1024);     // 1 MB

    gat_wh_kernel<<<N, 64, 0, stream>>>(h, W, a, src, dst, WhT);
    gat_attn_kernel<<<N / 32, 64, 0, stream>>>(adj, src, dst, WhT, out);
}
